// QuantLinear_77154792505750
// MI455X (gfx1250) — matrix operand order repaired and checked by reading, not compiled, not run on hardware
//
#include <hip/hip_runtime.h>

typedef __attribute__((ext_vector_type(16))) _Float16 v16h;
typedef __attribute__((ext_vector_type(8)))  _Float16 v8h;
typedef __attribute__((ext_vector_type(4)))  _Float16 v4h;
typedef __attribute__((ext_vector_type(2)))  _Float16 v2h;
typedef __attribute__((ext_vector_type(2)))  __fp16   v2hf;   // cvt_pkrtz return type
typedef __attribute__((ext_vector_type(8)))  float    v8f;
typedef __attribute__((ext_vector_type(4)))  int      v4i;

#define IN_F   4096
#define OUT_F  16384
#define BATCH  8
#define NBLK   (IN_F / 64)      // 64 absmax blocks per row
#define XPAD   8                // pad (halves) so A-row stride is 4 banks -> conflict-free
#define XROW   (IN_F + XPAD)    // halves per LDS row of x

__device__ const float NF4_LUT[16] = {
    -1.0f, -0.6961928009986877f, -0.5250730514526367f, -0.39491748809814453f,
    -0.28444138169288635f, -0.18477343022823334f, -0.09105003625154495f, 0.0f,
    0.07958029955625534f, 0.16093020141124725f, 0.24611230194568634f,
    0.33791524171829224f, 0.44070982933044434f, 0.5626170039176941f,
    0.7229568362236023f, 1.0f};

// pack two f32 into a f16 pair (v_cvt_pk_rtz_f16_f32)
__device__ __forceinline__ v2h pkh(float a, float b) {
  v2hf t = __builtin_amdgcn_cvt_pkrtz(a, b);
  return __builtin_bit_cast(v2h, t);
}

// dequantize 4 int32 codes -> 2 packed f16 pairs (absmax folded in f32)
__device__ __forceinline__ void dq4(v4i q, const float* __restrict__ lrep,
                                    float am, v2h& p0, v2h& p1) {
  p0 = pkh(lrep[q.x] * am, lrep[q.y] * am);
  p1 = pkh(lrep[q.z] * am, lrep[q.w] * am);
}

// dequantize 16 codes into one 16-bit B fragment (16 halves along K, one column)
__device__ __forceinline__ v16h dq16(v4i a, v4i b, v4i c, v4i d,
                                     const float* __restrict__ lrep, float am) {
  v2h p0, p1, p2, p3, p4, p5, p6, p7;
  dq4(a, lrep, am, p0, p1);
  dq4(b, lrep, am, p2, p3);
  dq4(c, lrep, am, p4, p5);
  dq4(d, lrep, am, p6, p7);
  v4h q0 = __builtin_shufflevector(p0, p1, 0, 1, 2, 3);
  v4h q1 = __builtin_shufflevector(p2, p3, 0, 1, 2, 3);
  v4h q2 = __builtin_shufflevector(p4, p5, 0, 1, 2, 3);
  v4h q3 = __builtin_shufflevector(p6, p7, 0, 1, 2, 3);
  v8h h0 = __builtin_shufflevector(q0, q1, 0, 1, 2, 3, 4, 5, 6, 7);
  v8h h1 = __builtin_shufflevector(q2, q3, 0, 1, 2, 3, 4, 5, 6, 7);
  return __builtin_shufflevector(h0, h1, 0, 1, 2, 3, 4, 5, 6, 7,
                                 8, 9, 10, 11, 12, 13, 14, 15);
}

__device__ __forceinline__ v16h aload(const _Float16* __restrict__ ap) {
  // A fragment: 16 contiguous halves at ap (contiguous K order, matches B)
  v8h lo = *(const v8h*)(ap);
  v8h hi = *(const v8h*)(ap + 8);
  return __builtin_shufflevector(lo, hi, 0, 1, 2, 3, 4, 5, 6, 7,
                                 8, 9, 10, 11, 12, 13, 14, 15);
}

__global__ __launch_bounds__(256, 1) void nf4_wmma_gemv(
    const float* __restrict__ x, const int* __restrict__ codes,
    const float* __restrict__ absmax, float* __restrict__ out) {
  // rows 0..7: x as f16; row 8: zeros (feeds A-matrix rows 8..15)
  __shared__ _Float16 xs[9 * XROW];
  __shared__ float lut[64];  // 4 replicas of 16-entry codebook (bank spreading)

  const int tid = threadIdx.x;
  if (tid < 64) lut[tid] = NF4_LUT[tid & 15];

  // stage x (8 x 4096 f32) -> LDS f16
  for (int i = tid * 4; i < BATCH * IN_F; i += 256 * 4) {
    const int row = i >> 12;
    const int col = i & (IN_F - 1);
    const float4 v = *(const float4*)(x + i);
    v2h a = pkh(v.x, v.y);
    v2h b = pkh(v.z, v.w);
    _Float16* dst = &xs[row * XROW + col];
    *(v2h*)(dst) = a;
    *(v2h*)(dst + 2) = b;
  }
  // zero row 8
  for (int i = tid * 2; i < XROW; i += 256 * 2) {
    v2h z = {};
    *(v2h*)(&xs[8 * XROW + i]) = z;
  }
  __syncthreads();

  const int lane = tid & 31;
  const int wave = tid >> 5;
  const int colN = lane & 15;     // B/D column, and A row for lanes 0..15
  const int hi   = lane >> 4;     // K-half select within fragments
  const int o    = (blockIdx.x * 8 + wave) * 16 + colN;   // output column
  const int mrow = (colN < BATCH) ? colN : BATCH;         // zero row for M>=8

  const _Float16* __restrict__ arow = &xs[mrow * XROW + hi * 16];
  const float*    __restrict__ lrep = &lut[(lane & 3) << 4];
  const int*      __restrict__ crow = codes + (size_t)o * IN_F + hi * 16;
  const float*    __restrict__ amr  = absmax + (size_t)o * NBLK;

  v8f acc = {};

  for (int blk = 0; blk < NBLK; ++blk) {
    const int kbase = blk * 64;
    const float am = amr[blk];
    const int* cp = crow + kbase;

    // stream this lane's 2x64B of codes (non-temporal: one-shot data)
    const v4i* cq = (const v4i*)cp;          // K = kbase + hi*16 + 0..15
    v4i q0 = __builtin_nontemporal_load(cq + 0);
    v4i q1 = __builtin_nontemporal_load(cq + 1);
    v4i q2 = __builtin_nontemporal_load(cq + 2);
    v4i q3 = __builtin_nontemporal_load(cq + 3);
    const v4i* cr = (const v4i*)(cp + 32);   // K = kbase + 32 + hi*16 + 0..15
    v4i r0 = __builtin_nontemporal_load(cr + 0);
    v4i r1 = __builtin_nontemporal_load(cr + 1);
    v4i r2 = __builtin_nontemporal_load(cr + 2);
    v4i r3 = __builtin_nontemporal_load(cr + 3);

    // prefetch next block's chunks
    __builtin_prefetch((const void*)(cp + 64), 0, 0);
    __builtin_prefetch((const void*)(cp + 96), 0, 0);

    // A fragments for K = kbase..+31 and kbase+32..+63 from LDS
    const _Float16* ap = arow + kbase;
    v16h a0 = aload(ap);
    v16h a1 = aload(ap + 32);

    // dequantize B fragments (LUT in LDS, absmax folded before f16 convert)
    v16h b0 = dq16(q0, q1, q2, q3, lrep, am);
    v16h b1 = dq16(r0, r1, r2, r3, lrep, am);

    acc = __builtin_amdgcn_wmma_f32_16x16x32_f16(false, a0, false, b0,
                                                 (short)0, acc, false, false);
    acc = __builtin_amdgcn_wmma_f32_16x16x32_f16(false, a1, false, b1,
                                                 (short)0, acc, false, false);
  }

  // C/D layout: VGPR r = row M=r for lanes 0..15 (rows 8..15 are zero-padding)
  if (lane < 16) {
#pragma unroll
    for (int r = 0; r < BATCH; ++r) {
      out[(size_t)r * OUT_F + o] = acc[r];
    }
  }
}

extern "C" void kernel_launch(void* const* d_in, const int* in_sizes, int n_in,
                              void* d_out, int out_size, void* d_ws,
                              size_t ws_size, hipStream_t stream) {
  (void)in_sizes; (void)n_in; (void)out_size; (void)d_ws; (void)ws_size;
  const float* x      = (const float*)d_in[0];
  const int*   codes  = (const int*)d_in[1];
  const float* absmax = (const float*)d_in[2];
  float*       out    = (float*)d_out;
  // 1024 wave-tiles of 16 columns, 8 waves (256 threads) per block
  nf4_wmma_gemv<<<dim3(OUT_F / 128), dim3(256), 0, stream>>>(x, codes, absmax,
                                                             out);
}
